// MTSCNNModel_25211458028029
// MI455X (gfx1250) — matrix operand order repaired and checked by reading, not compiled, not run on hardware
//
#include <hip/hip_runtime.h>

// ---------------------------------------------------------------------------
// MTS-CNN forward for MI455X (gfx1250, wave32, WMMA).
//
//   x[B,1,16,8,8] -> per-64-channel 1D CNN (conv 1->8 k3, leaky, pool2,
//   conv 8->16 k3, leaky, pool2) -> 32 feats/channel -> (linear FC stack,
//   collapsed) -> scalar per sample.
//
// Pipeline:
//   prep_weights : expand conv weights into WMMA-layout B matrices (f16) with
//                  bias folded into an extra K row; collapse the activation-
//                  free FC stack + detection head into Vt[c][16 oc][2 p].
//   prep_const   : scalar bias constant.
//   mtscnn_main  : per wave: 2 M-tiles (32 samples) x 64 channels of
//                  v_wmma_f32_16x16x32_f16, pooling via shfl_xor / v_max.
// ---------------------------------------------------------------------------

typedef __attribute__((ext_vector_type(16))) _Float16 v16h;
typedef __attribute__((ext_vector_type(8)))  _Float16 v8h;
typedef __attribute__((ext_vector_type(8)))  float    v8f;
typedef __attribute__((ext_vector_type(2)))  __fp16   fp16x2;  // cvt_pkrtz return type

#define NCHAN 64
#define NB    32768

// workspace layout (bytes)
#define W1EX_HALVES (NCHAN * 7 * 16 * 32)   // [c][j tile][n][k]  k fastest
#define W2EX_HALVES (NCHAN * 4 * 16 * 64)   // [c][j tile][n][k]
#define W1EX_OFF  0
#define W2EX_OFF  (W1EX_HALVES * 2)                       // 458752
#define VT_OFF    (W2EX_OFF + W2EX_HALVES * 2)            // 983040
#define VB_OFF    (VT_OFF + NCHAN * 32 * 4)               // 991232
#define BC_OFF    (VB_OFF + NCHAN * 4)                    // 991488

union V16H { v16h v; uint4 u4[2]; };

// ---------------------------------------------------------------------------
// Prep kernel: one block per channel.
// W1ex column order (per j-tile of 16): n<8 -> (oc=n, t=2j), n>=8 -> (oc=n-8,
// t=2j+1), so pool1 pairs are lane n and lane n^8. K row 16 carries the bias
// (A supplies 1.0 there). W2ex tile j == conv2 time-step t2=j, K = pooled1
// feature k2 = ic*7 + u (u = t2 + kk), K row 56 carries the bias.
// B-matrix per-lane layout follows the ISA 7.12.2 table order (same as A):
// lane l holds column n=l&15; element s<8 -> k = 8*(l>>4) + s,
// s>=8 -> k = 16 + 8*(l>>4) + (s-8), i.e. two 8-k octets per lane.
// ---------------------------------------------------------------------------
__global__ void prep_weights(const float* __restrict__ w1, const float* __restrict__ b1,
                             const float* __restrict__ w2, const float* __restrict__ b2,
                             const float* __restrict__ fw1, const float* __restrict__ fb1,
                             const float* __restrict__ fw2, const float* __restrict__ fb2,
                             const float* __restrict__ fw3, const float* __restrict__ fb3,
                             const float* __restrict__ dw1, const float* __restrict__ dw2,
                             _Float16* __restrict__ w1ex, _Float16* __restrict__ w2ex,
                             float* __restrict__ vt, float* __restrict__ vb) {
  const int c = blockIdx.x;
  const int tid = threadIdx.x;  // 128 threads

  // conv1 expanded weights: [7][16 n][32 k]
  for (int e = tid; e < 7 * 16 * 32; e += 128) {
    int k = e & 31, n = (e >> 5) & 15, j = e >> 9;
    int oc = n & 7, t = 2 * j + (n >> 3);
    float v = 0.0f;
    if (k < 16) {
      int kk = k - t;
      if (kk >= 0 && kk < 3) v = w1[(c * 8 + oc) * 3 + kk];
    } else if (k == 16) {
      v = b1[c * 8 + oc];
    }
    w1ex[(size_t)c * 3584 + e] = (_Float16)v;
  }

  // conv2 expanded weights: [4][16 n][64 k]
  for (int e = tid; e < 4 * 16 * 64; e += 128) {
    int k = e & 63, n = (e >> 6) & 15, j = e >> 10;
    float v = 0.0f;
    if (k < 56) {
      int ic = k / 7, u = k % 7, kk = u - j;
      if (kk >= 0 && kk < 3) v = w2[((c * 16 + n) * 8 + ic) * 3 + kk];
    } else if (k == 56) {
      v = b2[c * 16 + n];
    }
    w2ex[(size_t)c * 4096 + e] = (_Float16)v;
  }

  // collapse the linear FC stack: Vtilde = fw3_c * fw2_c * fw1_c  (1x32)
  __shared__ float t64[64];
  __shared__ float vtilde[32];
  __shared__ float vch;
  if (tid < 64) {
    float s = 0.0f;
    for (int o2 = 0; o2 < 32; ++o2) s += fw3[c * 32 + o2] * fw2[(c * 32 + o2) * 64 + tid];
    t64[tid] = s;
  }
  __syncthreads();
  if (tid < 32) {
    float s = 0.0f;
    for (int o1 = 0; o1 < 64; ++o1) s += t64[o1] * fw1[(c * 64 + o1) * 32 + tid];
    vtilde[tid] = s;
  }
  if (tid == 0) {
    float v = 0.0f;
    for (int o = 0; o < 32; ++o) v += dw2[o] * dw1[o * 64 + c];
    vch = v;
    float beta = 0.0f;
    for (int o2 = 0; o2 < 32; ++o2) {
      float s = fb2[c * 32 + o2];
      for (int o1 = 0; o1 < 64; ++o1) s += fw2[(c * 32 + o2) * 64 + o1] * fb1[c * 64 + o1];
      beta += fw3[c * 32 + o2] * s;
    }
    beta += fb3[c];
    vb[c] = v * beta;
  }
  __syncthreads();
  if (tid < 32) vt[c * 32 + tid] = vch * vtilde[tid];  // index f = oc*2 + p
}

__global__ void prep_const(const float* __restrict__ dw2, const float* __restrict__ db1,
                           const float* __restrict__ db2, const float* __restrict__ vb,
                           float* __restrict__ bc) {
  __shared__ float red[64];
  int t = threadIdx.x;
  float s = vb[t];
  if (t < 32) s += dw2[t] * db1[t];
  red[t] = s;
  __syncthreads();
  for (int st = 32; st > 0; st >>= 1) {
    if (t < st) red[t] += red[t + st];
    __syncthreads();
  }
  if (t == 0) *bc = red[0] + db2[0];
}

// ---------------------------------------------------------------------------
// Main kernel. 4 waves / 128 threads per WG; each wave owns 2 WMMA M-tiles
// (32 samples), WG covers 128 samples. x is staged 32 channels at a time into
// LDS as f16 [s][cslot][16 t] (131 KB) + per-wave pooled1 scratch (16 KB).
// ---------------------------------------------------------------------------
__global__ __launch_bounds__(128)
void mtscnn_main(const float* __restrict__ x, const _Float16* __restrict__ w1ex,
                 const _Float16* __restrict__ w2ex, const float* __restrict__ vt,
                 const float* __restrict__ bias_const, float* __restrict__ out) {
  extern __shared__ char smem[];
  _Float16* xs   = (_Float16*)smem;                        // [128][32][16]
  _Float16* pool = (_Float16*)(smem + 128 * 32 * 16 * 2);  // [8 mt][16 s][64 k2]

  const int tid  = threadIdx.x;
  const int lane = tid & 31;
  const int wave = tid >> 5;
  const int srow = lane & 15;        // A/B/C tile row-or-col owned by this lane
  const int hsel = lane >> 4;        // 0: K-octets {0..7,16..23}; 1: {8..15,24..31}
  const int wg_base = blockIdx.x * 128;

  // pooled1 pad region: k2=56 -> 1.0 (bias row of conv2 B), 57..63 -> 0.
  {
    int m = lane >> 4, row = lane & 15;
    uint4* d = (uint4*)(pool + (size_t)((wave * 2 + m) * 16 + row) * 64 + 56);
    d[0] = make_uint4(0x00003C00u, 0u, 0u, 0u);  // {1.0h, 0...}
  }

  float acc[2][8];
#pragma unroll
  for (int m = 0; m < 2; ++m)
#pragma unroll
    for (int r = 0; r < 8; ++r) acc[m][r] = 0.0f;

#pragma unroll 1
  for (int chalf = 0; chalf < 2; ++chalf) {
    __syncthreads();  // xs reuse barrier
    // ---- stage 32 channels x 128 samples, f32 -> f16, [t][c] -> [c][t] ----
#pragma unroll 1
    for (int i = 0; i < 32; ++i) {
      int p = tid + 128 * i;  // 0..4095
      int s = p >> 5, cs = p & 31;
      const float* gx = x + (size_t)(wg_base + s) * 1024 + (chalf * 32 + cs);
      uint32_t hb[8];
#pragma unroll
      for (int t2 = 0; t2 < 8; ++t2) {
        float a = gx[(2 * t2) * 64];
        float b = gx[(2 * t2 + 1) * 64];
        union { fp16x2 v; uint32_t u; } cv;
        cv.v = __builtin_amdgcn_cvt_pkrtz(a, b);
        hb[t2] = cv.u;
      }
      uint4* dst = (uint4*)(xs + ((size_t)s * 32 + cs) * 16);
      dst[0] = make_uint4(hb[0], hb[1], hb[2], hb[3]);
      dst[1] = make_uint4(hb[4], hb[5], hb[6], hb[7]);
    }
    __syncthreads();

    // ---- per-channel tiny CNN as 16x16x32 f16 WMMAs ----
#pragma unroll 1
    for (int cc = 0; cc < 32; ++cc) {
      const int c = chalf * 32 + cc;
      if (cc + 1 < 32) {  // warm L0/L2 for the next channel's B tiles
        __builtin_prefetch(w1ex + (size_t)(c + 1) * 3584, 0, 1);
        __builtin_prefetch(w2ex + (size_t)(c + 1) * 4096, 0, 1);
      }

      // A1 operand per M-tile: rows = samples, K = 16 time (+ K16 = 1.0 bias hook)
      V16H a1[2];
#pragma unroll
      for (int m = 0; m < 2; ++m) {
        const uint4* ap =
            (const uint4*)(xs + ((size_t)((wave * 2 + m) * 16 + srow) * 32 + cc) * 16 + hsel * 8);
        a1[m].u4[0] = ap[0];
        a1[m].u4[1] = make_uint4(hsel ? 0u : 0x00003C00u, 0u, 0u, 0u);
      }

      // conv1: 7 N-tiles of 16 cols (t=2j in lanes 0..7, t=2j+1 in 8..15)
#pragma unroll 1
      for (int j = 0; j < 7; ++j) {
        const _Float16* bp = w1ex + ((size_t)((c * 7 + j) * 16 + srow) * 32) + hsel * 8;
        V16H b1v;
        b1v.u4[0] = ((const uint4*)bp)[0];
        b1v.u4[1] = ((const uint4*)(bp + 16))[0];
#pragma unroll
        for (int m = 0; m < 2; ++m) {
          v8f cz = {};
          v8f c1 = __builtin_amdgcn_wmma_f32_16x16x32_f16(false, a1[m].v, false, b1v.v,
                                                          (short)0, cz, false, false);
#pragma unroll
          for (int r = 0; r < 8; ++r) {
            float xv = c1[r];
            float lv = fmaxf(xv, 0.05f * xv);              // leaky
            float pv = fmaxf(lv, __shfl_xor(lv, 8, 32));   // time-pair maxpool
            if ((lane & 8) == 0) {
              int sr = (wave * 2 + m) * 16 + r + ((lane & 16) ? 8 : 0);
              pool[(size_t)sr * 64 + (lane & 7) * 7 + j] = (_Float16)pv;
            }
          }
        }
      }

      // A2 operands: pooled1 features k2=0..55 (+bias 1.0 at 56), K split 0..31 / 32..63
      V16H a2a[2], a2b[2];
#pragma unroll
      for (int m = 0; m < 2; ++m) {
        const _Float16* pb = pool + (size_t)((wave * 2 + m) * 16 + srow) * 64;
        a2a[m].u4[0] = ((const uint4*)(pb + hsel * 8))[0];
        a2a[m].u4[1] = ((const uint4*)(pb + 16 + hsel * 8))[0];
        a2b[m].u4[0] = ((const uint4*)(pb + 32 + hsel * 8))[0];
        a2b[m].u4[1] = ((const uint4*)(pb + 48 + hsel * 8))[0];
      }
      const float2 vt2 = *(const float2*)(vt + c * 32 + srow * 2);

      // conv2: N-tile j == time-step t2 (t2=4 is trimmed by the pool -> skipped)
#pragma unroll
      for (int jj = 0; jj < 2; ++jj) {
        const _Float16* bpA = w2ex + ((size_t)((c * 4 + 2 * jj) * 16 + srow) * 64) + hsel * 8;
        const _Float16* bpB = w2ex + ((size_t)((c * 4 + 2 * jj + 1) * 16 + srow) * 64) + hsel * 8;
        V16H bA0, bA1, bB0, bB1;
        bA0.u4[0] = ((const uint4*)bpA)[0];        bA0.u4[1] = ((const uint4*)(bpA + 16))[0];
        bA1.u4[0] = ((const uint4*)(bpA + 32))[0]; bA1.u4[1] = ((const uint4*)(bpA + 48))[0];
        bB0.u4[0] = ((const uint4*)bpB)[0];        bB0.u4[1] = ((const uint4*)(bpB + 16))[0];
        bB1.u4[0] = ((const uint4*)(bpB + 32))[0]; bB1.u4[1] = ((const uint4*)(bpB + 48))[0];
        const float w = jj ? vt2.y : vt2.x;
#pragma unroll
        for (int m = 0; m < 2; ++m) {
          v8f z = {};
          v8f cA = __builtin_amdgcn_wmma_f32_16x16x32_f16(false, a2a[m].v, false, bA0.v,
                                                          (short)0, z, false, false);
          cA = __builtin_amdgcn_wmma_f32_16x16x32_f16(false, a2b[m].v, false, bA1.v,
                                                      (short)0, cA, false, false);
          v8f cB = __builtin_amdgcn_wmma_f32_16x16x32_f16(false, a2a[m].v, false, bB0.v,
                                                          (short)0, z, false, false);
          cB = __builtin_amdgcn_wmma_f32_16x16x32_f16(false, a2b[m].v, false, bB1.v,
                                                      (short)0, cB, false, false);
#pragma unroll
          for (int r = 0; r < 8; ++r) {
            float lA = fmaxf(cA[r], 0.05f * cA[r]);
            float lB = fmaxf(cB[r], 0.05f * cB[r]);
            acc[m][r] = fmaf(fmaxf(lA, lB), w, acc[m][r]);  // pool2 + collapsed FC
          }
        }
      }
    }
  }

  // ---- reduce feature columns across 16 lanes, write one f32 per sample ----
  const float bc = *bias_const;
#pragma unroll
  for (int m = 0; m < 2; ++m)
#pragma unroll
    for (int r = 0; r < 8; ++r) {
      float v = acc[m][r];
      v += __shfl_xor(v, 1, 32);
      v += __shfl_xor(v, 2, 32);
      v += __shfl_xor(v, 4, 32);
      v += __shfl_xor(v, 8, 32);
      if ((lane & 15) == 0) {
        int sidx = wg_base + (wave * 2 + m) * 16 + r + ((lane & 16) ? 8 : 0);
        out[sidx] = v + bc;
      }
    }
}

extern "C" void kernel_launch(void* const* d_in, const int* in_sizes, int n_in,
                              void* d_out, int out_size, void* d_ws, size_t ws_size,
                              hipStream_t stream) {
  (void)in_sizes; (void)n_in; (void)out_size; (void)ws_size;
  const float* x   = (const float*)d_in[0];
  const float* w1  = (const float*)d_in[1];
  const float* b1  = (const float*)d_in[2];
  const float* w2  = (const float*)d_in[3];
  const float* b2  = (const float*)d_in[4];
  const float* fw1 = (const float*)d_in[5];
  const float* fb1 = (const float*)d_in[6];
  const float* fw2 = (const float*)d_in[7];
  const float* fb2 = (const float*)d_in[8];
  const float* fw3 = (const float*)d_in[9];
  const float* fb3 = (const float*)d_in[10];
  const float* dw1 = (const float*)d_in[11];
  const float* db1 = (const float*)d_in[12];
  const float* dw2 = (const float*)d_in[13];
  const float* db2 = (const float*)d_in[14];

  char* ws = (char*)d_ws;
  _Float16* w1ex = (_Float16*)(ws + W1EX_OFF);
  _Float16* w2ex = (_Float16*)(ws + W2EX_OFF);
  float*    vt   = (float*)(ws + VT_OFF);
  float*    vb   = (float*)(ws + VB_OFF);
  float*    bc   = (float*)(ws + BC_OFF);

  prep_weights<<<NCHAN, 128, 0, stream>>>(w1, b1, w2, b2, fw1, fb1, fw2, fb2, fw3, fb3,
                                          dw1, dw2, w1ex, w2ex, vt, vb);
  prep_const<<<1, 64, 0, stream>>>(dw2, db1, db2, vb, bc);

  const size_t lds_bytes = 128 * 32 * 16 * 2   /* x stage, f16 */
                         + 8 * 16 * 64 * 2;    /* pooled1       */
  mtscnn_main<<<NB / 128, 128, lds_bytes, stream>>>(x, w1ex, w2ex, vt, bc, (float*)d_out);
}
